// PointEmbed_30296699306682
// MI455X (gfx1250) — matrix operand order repaired and checked by reading, not compiled, not run on hardware
//
#include <hip/hip_runtime.h>
#include <hip/hip_bf16.h>
#include <stdint.h>

// ---------- types ----------
typedef _Float16 vhalf16 __attribute__((ext_vector_type(16)));
typedef _Float16 vhalf8  __attribute__((ext_vector_type(8)));
typedef float    vfloat8 __attribute__((ext_vector_type(8)));

#define EPSF 1e-5f
#define BATCH 8
#define NPTS 8192
#define NPOINT 1024
#define NSAMPLE 32

// output layout (floats)
#define OFF_XYZ0  ((size_t)0)
#define OFF_FEAT0 ((size_t)(BATCH*NPOINT*3))                       // 24576
#define OFF_XYZ1  (OFF_FEAT0 + (size_t)BATCH*128*NPOINT)           // 1073152
#define OFF_FEAT1 (OFF_XYZ1 + (size_t)BATCH*NPOINT*3)              // 1097728

// ---------- WMMA fragment loaders (wave32, contiguous k order used by both A and B) ----------
// A-matrix 16x32 f16: lane l holds row M=l&15; halves 0..15 -> K = k0 + 16*(l>=16) + h.
__device__ __forceinline__ vhalf16 load_a_frag(const _Float16* base, int stride,
                                               int k0, int lane) {
  int mrow = lane & 15;
  int kb   = k0 + ((lane >> 4) << 4);
  const _Float16* p = base + mrow * stride + kb;
  vhalf8 lo = *(const vhalf8*)(p);
  vhalf8 hi = *(const vhalf8*)(p + 8);
  return __builtin_shufflevector(lo, hi, 0,1,2,3,4,5,6,7,8,9,10,11,12,13,14,15);
}

// B-matrix 32x16 f16 (B[k][n] = W[n][k], W row-major NxK):
// lanes 0-15: N=n0+lane, K = k0..k0+15 across halves; lanes 16-31: same N, K += 16.
__device__ __forceinline__ vhalf16 load_b_frag(const _Float16* w, int stride,
                                               int n0, int k0, int lane) {
  int n  = n0 + (lane & 15);
  int kb = k0 + ((lane >> 4) << 4);
  return *(const vhalf16*)(w + n * stride + kb);
}

// ---------- kernel 0: weight f32 -> f16 ----------
__global__ void prep_kernel(const float* __restrict__ w2,
                            const float* __restrict__ wl1,
                            const float* __restrict__ wl2,
                            _Float16* __restrict__ w2h,
                            _Float16* __restrict__ wl1h,
                            _Float16* __restrict__ wl2h) {
  int t = blockIdx.x * 256 + threadIdx.x;
  if (t < 64 * 64)  w2h[t]  = (_Float16)w2[t];
  if (t < 128 * 128) { wl1h[t] = (_Float16)wl1[t]; wl2h[t] = (_Float16)wl2[t]; }
}

// ---------- kernel 1: conv1 (scalar, K=3) + conv2 (WMMA f16) ----------
// block = 256 threads (8 waves), 128 points/block, 512 blocks
__global__ void embed_kernel(const float* __restrict__ x,
                             const float* __restrict__ w1,      // 64x3
                             const _Float16* __restrict__ w2h,  // 64x64
                             const float* __restrict__ g1, const float* __restrict__ b1,
                             const float* __restrict__ m1, const float* __restrict__ v1,
                             const float* __restrict__ g2, const float* __restrict__ b2,
                             const float* __restrict__ m2, const float* __restrict__ v2,
                             _Float16* __restrict__ fh) {      // (B*N, 64)
  __shared__ __align__(32) _Float16 f1[128 * 64];
  __shared__ float sx[128 * 3];
  const int base = blockIdx.x * 128;

  for (int t = threadIdx.x; t < 128 * 3; t += 256)
    sx[t] = x[(size_t)base * 3 + t];
  __syncthreads();

  // stage 1: 3 -> 64, bn1, relu, f16 into LDS
  for (int t = threadIdx.x; t < 128 * 64; t += 256) {
    int p = t >> 6, c = t & 63;
    float val = sx[p * 3 + 0] * w1[c * 3 + 0] +
                sx[p * 3 + 1] * w1[c * 3 + 1] +
                sx[p * 3 + 2] * w1[c * 3 + 2];
    float sc = g1[c] * rsqrtf(v1[c] + EPSF);
    val = (val - m1[c]) * sc + b1[c];
    f1[t] = (_Float16)fmaxf(val, 0.0f);
  }
  __syncthreads();

  // stage 2: 64 -> 64 via WMMA, each wave owns a 16-point tile
  const int wave = threadIdx.x >> 5, lane = threadIdx.x & 31;
  const _Float16* atile = f1 + wave * 16 * 64;
  vhalf16 a0 = load_a_frag(atile, 64, 0, lane);
  vhalf16 a1 = load_a_frag(atile, 64, 32, lane);
  const int mcol = lane & 15;
  const int rowo = (lane >> 4) * 8;

#pragma unroll
  for (int n0 = 0; n0 < 64; n0 += 16) {
    vfloat8 acc = {};
    vhalf16 bf0 = load_b_frag(w2h, 64, n0, 0, lane);
    vhalf16 bf1 = load_b_frag(w2h, 64, n0, 32, lane);
    acc = __builtin_amdgcn_wmma_f32_16x16x32_f16(false, a0, false, bf0,
                                                 (short)0, acc, false, false);
    acc = __builtin_amdgcn_wmma_f32_16x16x32_f16(false, a1, false, bf1,
                                                 (short)0, acc, false, false);
    int ch = n0 + mcol;
    float sc = g2[ch] * rsqrtf(v2[ch] + EPSF);
    float bi = b2[ch] - m2[ch] * sc;
#pragma unroll
    for (int i = 0; i < 8; ++i) {
      float val = fmaxf(acc[i] * sc + bi, 0.0f);
      int pt = base + wave * 16 + rowo + i;
      fh[(size_t)pt * 64 + ch] = (_Float16)val;
    }
  }
}

// ---------- kernel 2: farthest point sampling ----------
// 8 blocks x 1024 threads; each thread owns 8 points
__global__ void fps_kernel(const float* __restrict__ x,
                           int* __restrict__ fps_idx,
                           float* __restrict__ out) {
  const int b = blockIdx.x;
  const int tid = threadIdx.x;
  const float* xb = x + (size_t)b * NPTS * 3;

  float dist[8], px[8], py[8], pz[8];
#pragma unroll
  for (int j = 0; j < 8; ++j) {
    int p = tid + j * 1024;
    px[j] = xb[p * 3 + 0]; py[j] = xb[p * 3 + 1]; pz[j] = xb[p * 3 + 2];
    dist[j] = 1e10f;
  }

  __shared__ float s_val[32];
  __shared__ int   s_idx[32];
  __shared__ int   s_far;
  __shared__ int   s_fidx[NPOINT];

  int far = 0;
  for (int iter = 0; iter < NPOINT; ++iter) {
    if (tid == 0) s_fidx[iter] = far;   // record *incoming* far (matches scan)
    float cx = xb[far * 3 + 0], cy = xb[far * 3 + 1], cz = xb[far * 3 + 2];
    float best = -1.0f; int bidx = tid;
#pragma unroll
    for (int j = 0; j < 8; ++j) {
      float dx = px[j] - cx, dy = py[j] - cy, dz = pz[j] - cz;
      float d = dx * dx + dy * dy + dz * dz;
      dist[j] = fminf(dist[j], d);
      if (dist[j] > best) { best = dist[j]; bidx = tid + j * 1024; }
    }
#pragma unroll
    for (int off = 16; off > 0; off >>= 1) {
      float ov = __shfl_xor(best, off, 32);
      int   oi = __shfl_xor(bidx, off, 32);
      if (ov > best || (ov == best && oi < bidx)) { best = ov; bidx = oi; }
    }
    int wv = tid >> 5;
    if ((tid & 31) == 0) { s_val[wv] = best; s_idx[wv] = bidx; }
    __syncthreads();
    if (wv == 0) {
      best = s_val[tid & 31]; bidx = s_idx[tid & 31];
#pragma unroll
      for (int off = 16; off > 0; off >>= 1) {
        float ov = __shfl_xor(best, off, 32);
        int   oi = __shfl_xor(bidx, off, 32);
        if (ov > best || (ov == best && oi < bidx)) { best = ov; bidx = oi; }
      }
      if (tid == 0) s_far = bidx;
    }
    __syncthreads();
    far = s_far;
  }
  __syncthreads();

  // emit fps_idx and both new_xyz copies
  int idx = s_fidx[tid];
  fps_idx[b * NPOINT + tid] = idx;
#pragma unroll
  for (int k = 0; k < 3; ++k) {
    float v = xb[idx * 3 + k];
    size_t o = (size_t)(b * NPOINT + tid) * 3 + k;
    out[OFF_XYZ0 + o] = v;
    out[OFF_XYZ1 + o] = v;
  }
}

// ---------- kernel 3: 32-NN per sampled point ----------
// 8192 blocks (one per query) x 256 threads; full distance row in LDS
__global__ void knn_kernel(const float* __restrict__ x,
                           const float* __restrict__ out,   // new_xyz at OFF_XYZ0
                           int* __restrict__ knn_idx) {
  __shared__ float d2[NPTS];
  __shared__ float s_val[8];
  __shared__ int   s_idx[8];
  const int q = blockIdx.x;
  const int b = q >> 10, s = q & 1023;
  const float* xb = x + (size_t)b * NPTS * 3;
  const float* nq = out + OFF_XYZ0 + (size_t)(b * NPOINT + s) * 3;
  const float qx = nq[0], qy = nq[1], qz = nq[2];

  for (int p = threadIdx.x; p < NPTS; p += 256) {
    float dx = xb[p * 3 + 0] - qx, dy = xb[p * 3 + 1] - qy, dz = xb[p * 3 + 2] - qz;
    d2[p] = dx * dx + dy * dy + dz * dz;
  }
  __syncthreads();

  const int lane = threadIdx.x & 31, wv = threadIdx.x >> 5;
  for (int r = 0; r < NSAMPLE; ++r) {
    float best = 3.4e38f; int bidx = 0;
    for (int p = threadIdx.x; p < NPTS; p += 256) {
      float d = d2[p];
      if (d < best) { best = d; bidx = p; }
    }
#pragma unroll
    for (int off = 16; off > 0; off >>= 1) {
      float ov = __shfl_xor(best, off, 32);
      int   oi = __shfl_xor(bidx, off, 32);
      if (ov < best || (ov == best && oi < bidx)) { best = ov; bidx = oi; }
    }
    if (lane == 0) { s_val[wv] = best; s_idx[wv] = bidx; }
    __syncthreads();
    if (threadIdx.x == 0) {
      float fb = s_val[0]; int fi = s_idx[0];
#pragma unroll
      for (int w2 = 1; w2 < 8; ++w2) {
        if (s_val[w2] < fb || (s_val[w2] == fb && s_idx[w2] < fi)) {
          fb = s_val[w2]; fi = s_idx[w2];
        }
      }
      knn_idx[(size_t)q * NSAMPLE + r] = fi;
      d2[fi] = 3.4e38f;
    }
    __syncthreads();
  }
}

// ---------- kernel 4: group + lconv1 + lconv2 + max pool (WMMA-heavy) ----------
// 2048 blocks x 256 threads (8 waves); 4 queries/block, 2 waves (M-tiles)/query.
// feat is overwritten in place by h: each wave touches only its own 16-row tile,
// and its A fragments are register-resident before any LDS store (LDS is in-order
// per wave), so no extra barrier is required between GEMM1 and GEMM2.
__global__ void group_lconv_kernel(const _Float16* __restrict__ fh,
                                   const int* __restrict__ fps_idx,
                                   const int* __restrict__ knn_idx,
                                   const _Float16* __restrict__ w1,  // 128x128
                                   const _Float16* __restrict__ w2,  // 128x128
                                   const float* __restrict__ g1, const float* __restrict__ b1,
                                   const float* __restrict__ m1, const float* __restrict__ v1,
                                   const float* __restrict__ g2, const float* __restrict__ b2,
                                   const float* __restrict__ m2, const float* __restrict__ v2,
                                   float* __restrict__ out) {
  __shared__ __align__(32) _Float16 feat[4][32][128];   // reused as h after GEMM1
  __shared__ _Float16 np[4][64];
  __shared__ float part[4][2][128];
  const int qbase = blockIdx.x * 4;

  // new_pts gather (f16): 4 queries x 64 ch = 256 entries
  {
    int q = threadIdx.x >> 6, c = threadIdx.x & 63;
    int gq = qbase + q;
    int b = gq >> 10, s = gq & 1023;
    int fidx = fps_idx[b * NPOINT + s];
    np[q][c] = fh[(size_t)(b * NPTS + fidx) * 64 + c];
  }
  __syncthreads();

  // build feat = [grouped - new_pts , new_pts]  (4 q x 32 rows = 128 rows)
  if (threadIdx.x < 128) {
    int q = threadIdx.x >> 5, j = threadIdx.x & 31;
    int gq = qbase + q;
    int b = gq >> 10;
    int nbr = knn_idx[(size_t)gq * NSAMPLE + j];
    const _Float16* fp = fh + (size_t)(b * NPTS + nbr) * 64;
#pragma unroll 8
    for (int c = 0; c < 64; ++c) {
      _Float16 npc = np[q][c];
      feat[q][j][c]      = (_Float16)((float)fp[c] - (float)npc);
      feat[q][j][64 + c] = npc;
    }
  }
  __syncthreads();

  const int wave = threadIdx.x >> 5, lane = threadIdx.x & 31;
  const int q = wave >> 1, mt = wave & 1;
  const int mcol = lane & 15, rowo = (lane >> 4) * 8;
  _Float16* tile = &feat[q][mt * 16][0];   // wave-private 16x128 tile

  // GEMM1: feat(16x128) @ W1^T -> bn -> relu -> overwrite tile with h
  {
    vhalf16 a[4];
#pragma unroll
    for (int k = 0; k < 4; ++k) a[k] = load_a_frag(tile, 128, k * 32, lane);
#pragma unroll 2
    for (int n0 = 0; n0 < 128; n0 += 16) {
      vfloat8 acc = {};
#pragma unroll
      for (int k = 0; k < 4; ++k) {
        vhalf16 bf = load_b_frag(w1, 128, n0, k * 32, lane);
        acc = __builtin_amdgcn_wmma_f32_16x16x32_f16(false, a[k], false, bf,
                                                     (short)0, acc, false, false);
      }
      int ch = n0 + mcol;
      float sc = g1[ch] * rsqrtf(v1[ch] + EPSF);
      float bi = b1[ch] - m1[ch] * sc;
#pragma unroll
      for (int i = 0; i < 8; ++i)
        tile[(rowo + i) * 128 + ch] = (_Float16)fmaxf(acc[i] * sc + bi, 0.0f);
    }
  }
  // no barrier: each wave reads back only its own tile (LDS in-order per wave)

  // GEMM2: h(16x128) @ W2^T -> bn -> relu -> max over rows
  {
    vhalf16 a[4];
#pragma unroll
    for (int k = 0; k < 4; ++k) a[k] = load_a_frag(tile, 128, k * 32, lane);
#pragma unroll 2
    for (int n0 = 0; n0 < 128; n0 += 16) {
      vfloat8 acc = {};
#pragma unroll
      for (int k = 0; k < 4; ++k) {
        vhalf16 bf = load_b_frag(w2, 128, n0, k * 32, lane);
        acc = __builtin_amdgcn_wmma_f32_16x16x32_f16(false, a[k], false, bf,
                                                     (short)0, acc, false, false);
      }
      int ch = n0 + mcol;
      float sc = g2[ch] * rsqrtf(v2[ch] + EPSF);
      float bi = b2[ch] - m2[ch] * sc;
      float mx = -3.4e38f;
#pragma unroll
      for (int i = 0; i < 8; ++i)
        mx = fmaxf(mx, fmaxf(acc[i] * sc + bi, 0.0f));
      mx = fmaxf(mx, __shfl_xor(mx, 16, 32));   // combine rows 0-7 with 8-15
      if (lane < 16) part[q][mt][ch] = mx;
    }
  }
  __syncthreads();

  // combine the two 16-row tiles, transpose-store feature_0 (B,128,S), both copies
  for (int t = threadIdx.x; t < 512; t += 256) {
    int q2 = t >> 7, ch = t & 127;
    int gq = qbase + q2;
    int b = gq >> 10, s = gq & 1023;
    float val = fmaxf(part[q2][0][ch], part[q2][1][ch]);
    size_t o = ((size_t)b * 128 + ch) * NPOINT + s;
    out[OFF_FEAT0 + o] = val;
    out[OFF_FEAT1 + o] = val;
  }
}

// ---------- host ----------
extern "C" void kernel_launch(void* const* d_in, const int* in_sizes, int n_in,
                              void* d_out, int out_size, void* d_ws, size_t ws_size,
                              hipStream_t stream) {
  const float* x   = (const float*)d_in[0];
  const float* w1  = (const float*)d_in[1];   // conv1_w 64x3
  const float* w2  = (const float*)d_in[2];   // conv2_w 64x64
  const float* wl1 = (const float*)d_in[3];   // lconv1_w 128x128
  const float* wl2 = (const float*)d_in[4];   // lconv2_w 128x128
  const float* bn1g = (const float*)d_in[5],  *bn1b = (const float*)d_in[6];
  const float* bn1m = (const float*)d_in[7],  *bn1v = (const float*)d_in[8];
  const float* bn2g = (const float*)d_in[9],  *bn2b = (const float*)d_in[10];
  const float* bn2m = (const float*)d_in[11], *bn2v = (const float*)d_in[12];
  const float* lb1g = (const float*)d_in[13], *lb1b = (const float*)d_in[14];
  const float* lb1m = (const float*)d_in[15], *lb1v = (const float*)d_in[16];
  const float* lb2g = (const float*)d_in[17], *lb2b = (const float*)d_in[18];
  const float* lb2m = (const float*)d_in[19], *lb2v = (const float*)d_in[20];
  float* out = (float*)d_out;

  // workspace carve-up (256B-aligned offsets)
  char* ws = (char*)d_ws;
  _Float16* fh    = (_Float16*)(ws);                       // 8*8192*64*2 = 8388608
  _Float16* w2h   = (_Float16*)(ws + 8388608);             // 8192
  _Float16* wl1h  = (_Float16*)(ws + 8388608 + 8192);      // 32768
  _Float16* wl2h  = (_Float16*)(ws + 8388608 + 40960);     // 32768
  int*      fidx  = (int*)     (ws + 8388608 + 73728);     // 32768
  int*      kidx  = (int*)     (ws + 8388608 + 106496);    // 1048576

  prep_kernel<<<64, 256, 0, stream>>>(w2, wl1, wl2, w2h, wl1h, wl2h);

  embed_kernel<<<BATCH * NPTS / 128, 256, 0, stream>>>(
      x, w1, w2h, bn1g, bn1b, bn1m, bn1v, bn2g, bn2b, bn2m, bn2v, fh);

  fps_kernel<<<BATCH, 1024, 0, stream>>>(x, fidx, out);

  knn_kernel<<<BATCH * NPOINT, 256, 0, stream>>>(x, out, kidx);

  group_lconv_kernel<<<BATCH * NPOINT / 4, 256, 0, stream>>>(
      fh, fidx, kidx, wl1h, wl2h,
      lb1g, lb1b, lb1m, lb1v, lb2g, lb2b, lb2m, lb2v, out);
}
